// MemoryGame_91293824844289
// MI455X (gfx1250) — matrix operand order repaired and checked by reading, not compiled, not run on hardware
//
#include <hip/hip_runtime.h>
#include <hip/hip_bf16.h>

// ---------------- problem constants (match reference) ----------------
constexpr int NX = 128;
constexpr int NG = 64;
constexpr int P  = NX * NG;      // 8192
constexpr int T  = 4;
constexpr int N_ITER = 50;
constexpr float KAPPA = 0.8f;
constexpr float LAMDA = 0.9f;
constexpr float YITA  = 0.1f;
constexpr float SLOPE = 0.01f;   // LeakyReLU negative slope

constexpr int KSPLIT = 8;            // K-reduction split across blocks
constexpr int KCHUNK = P / KSPLIT;   // 1024 rows per block

typedef unsigned short u16;
typedef __attribute__((ext_vector_type(16))) __bf16  v16bf;
typedef __attribute__((ext_vector_type(8)))  float   v8f;
typedef __attribute__((ext_vector_type(8)))  unsigned int v8u;
typedef __attribute__((ext_vector_type(4)))  int     v4i;

#if defined(__gfx1250__) && __has_builtin(__builtin_amdgcn_global_load_async_to_lds_b128)
#define ASYNC_LDS 1
#else
#define ASYNC_LDS 0
#endif

// ---------------- helpers ----------------
__device__ __forceinline__ float f_p(float v) {
    float c = fminf(fmaxf(v, -1.0f), 1.0f);
    return (c >= 0.0f) ? c : SLOPE * c;
}
__device__ __forceinline__ float bf2f(unsigned int lo16) {
    unsigned int x = lo16 << 16;
    return __builtin_bit_cast(float, x);
}
__device__ __forceinline__ u16 f2bf(float f) {
    __bf16 b = (__bf16)f;           // RNE convert
    return __builtin_bit_cast(u16, b);
}
__device__ __forceinline__ unsigned int pair_update(unsigned int mm, float ui,
                                                    float va, float vb) {
    float a = LAMDA * bf2f(mm & 0xffffu) + ui * va;
    float b = LAMDA * bf2f(mm >> 16)     + ui * vb;
    return (unsigned int)f2bf(a) | ((unsigned int)f2bf(b) << 16);
}

// 16B global -> LDS, async when available (ASYNCcnt path), else via VGPRs
__device__ __forceinline__ void tile_ld16(const u16* g, u16* l) {
#if ASYNC_LDS
    __builtin_amdgcn_global_load_async_to_lds_b128(
        (__attribute__((address_space(1))) v4i*)(u16*)g,
        (__attribute__((address_space(3))) v4i*)l, 0, 0);
#else
    uint4 d = *(const uint4*)g;
    uint2* s = (uint2*)l;
    uint2 a; a.x = d.x; a.y = d.y;
    uint2 b; b.x = d.z; b.y = d.w;
    s[0] = a; s[1] = b;
#endif
}
__device__ __forceinline__ void tile_ld_wait() {
#if ASYNC_LDS
#if __has_builtin(__builtin_amdgcn_s_wait_asynccnt)
    __builtin_amdgcn_s_wait_asynccnt(0);
#else
    asm volatile("s_wait_asynccnt 0" ::: "memory");
#endif
#endif
}

// ---------------- M0 (fp32) -> M bf16 (L2-resident working copy) ----------------
__global__ __launch_bounds__(256) void k_cvt(const float* __restrict__ M0,
                                             u16* __restrict__ Mb) {
    size_t i8   = (size_t)blockIdx.x * blockDim.x + threadIdx.x; // 8 elems/thread
    size_t base = i8 * 8;
    const float4* s = (const float4*)(M0 + base);
    float4 a = s[0], b = s[1];
    uint4 o;
    o.x = (unsigned)f2bf(a.x) | ((unsigned)f2bf(a.y) << 16);
    o.y = (unsigned)f2bf(a.z) | ((unsigned)f2bf(a.w) << 16);
    o.z = (unsigned)f2bf(b.x) | ((unsigned)f2bf(b.y) << 16);
    o.w = (unsigned)f2bf(b.z) | ((unsigned)f2bf(b.w) << 16);
    *(uint4*)(Mb + base) = o;
}

// ---------------- h0 = f_p(tiled g) ----------------
__global__ __launch_bounds__(256) void k_init_h(const float* __restrict__ g, int t,
                                                float* __restrict__ h,
                                                u16* __restrict__ hbf) {
    int j = blockIdx.x * blockDim.x + threadIdx.x;
    float v = f_p(g[t * NG + (j & (NG - 1))]);
    h[j]   = v;
    hbf[j] = f2bf(v);
}

// ---------------- y = h @ M via WMMA (bf16, M resident in L2) ----------------
// Grid: (P/128) x KSPLIT blocks of 256 threads (8 waves).
// blockIdx.x -> 128 output columns; blockIdx.y -> K-range of 1024.
// Double-buffered LDS; next tile streamed with async global->LDS loads
// (ASYNCcnt) while the wave runs v_wmma_f32_16x16x32_bf16 on the current tile.
constexpr int LROW = 132;   // padded LDS row stride (bf16): 8*132 % 64 != 0
__global__ __launch_bounds__(256) void k_gemv(const u16* __restrict__ Mb,
                                              const u16* __restrict__ hbf,
                                              float* __restrict__ ypart) {
    __shared__ u16 tile[2][32 * LROW];
    const int tid  = threadIdx.x;
    const int lane = tid & 31;
    const int wv   = tid >> 5;
    const int jb   = blockIdx.x * 128;
    const int kbeg = blockIdx.y * KCHUNK;
    const int rr   = tid >> 4;          // staging row 0..15
    const int cc   = tid & 15;          // staging chunk (8 bf16)
    const int half = lane >> 4;         // 0: K 0..15 side, 1: K 16..31 side
    const int colL = wv * 16 + (lane & 15);
    const unsigned am = ((lane & 15) == 0) ? 0xFFFFFFFFu : 0u; // A row-0 mask

    const u16* gptr = Mb + (size_t)(kbeg + rr) * P + jb + cc * 8;
    const u16* hptr = hbf + kbeg + half * 16;
    const int loff0 = rr * LROW + cc * 8;
    const int loff1 = (rr + 16) * LROW + cc * 8;

    // ---- prologue: stage tile 0 ----
    tile_ld16(gptr,                  &tile[0][loff0]);
    tile_ld16(gptr + (size_t)16 * P, &tile[0][loff1]);
    gptr += (size_t)32 * P;
    tile_ld_wait();

    v8f acc = {};
    int cur = 0;
    for (int it = 0; it < KCHUNK / 32; ++it) {
        __syncthreads();   // tile[cur] staged; prior reads of tile[cur^1] done

        // ---- stream next tile into the other buffer (overlaps compute) ----
        // (last iter reads the pad region; data never consumed)
        tile_ld16(gptr,                  &tile[cur ^ 1][loff0]);
        tile_ld16(gptr + (size_t)16 * P, &tile[cur ^ 1][loff1]);
        gptr += (size_t)32 * P;

        // ---- A fragment: row 0 = h[k0..k0+32), rows 1..15 zero (branchless) ----
        uint4 A0 = *(const uint4*)hptr;          // A elems 0..7
        uint4 A1 = *(const uint4*)(hptr + 8);    // A elems 8..15
        hptr += 32;
        v8u ap = {A0.x & am, A0.y & am, A0.z & am, A0.w & am,
                  A1.x & am, A1.y & am, A1.z & am, A1.w & am};
        v16bf af = __builtin_bit_cast(v16bf, ap);

        // ---- B fragment: column colL, K = half*16 + e ----
        v16bf bfv;
        const u16* tsrc = &tile[cur][(half * 16) * LROW + colL];
        #pragma unroll
        for (int e = 0; e < 16; ++e)
            bfv[e] = __builtin_bit_cast(__bf16, tsrc[e * LROW]);

        acc = __builtin_amdgcn_wmma_f32_16x16x32_bf16(
            false, af, false, bfv, (short)0, acc, false, false);

        tile_ld_wait();    // next tile committed before we pass the next barrier
        cur ^= 1;
    }
    // C/D layout: VGPR0 of lanes 0..15 = (M=0, N=lane) -> row 0 = partial y
    if (half == 0) ypart[blockIdx.y * P + jb + colL] = acc[0];
}

// ---------------- h = f_p(kappa*h + h*sum_k ypart) ----------------
__global__ __launch_bounds__(256) void k_update_h(float* __restrict__ h,
                                                  u16* __restrict__ hbf,
                                                  const float* __restrict__ ypart) {
    int j = blockIdx.x * blockDim.x + threadIdx.x;
    float yv = 0.0f;
    #pragma unroll
    for (int ks = 0; ks < KSPLIT; ++ks) yv += ypart[ks * P + j];
    float hv = h[j];
    float nv = f_p(KAPPA * hv + hv * yv);
    h[j]   = nv;
    hbf[j] = f2bf(nv);
}

// ---------------- loss += sum|p-h|; u = p+h; v = yita*(p-h) ----------------
__global__ __launch_bounds__(256) void k_finalize(const float* __restrict__ x,
                                                  const float* __restrict__ g, int t,
                                                  const float* __restrict__ h,
                                                  float* __restrict__ u,
                                                  float* __restrict__ v,
                                                  float* __restrict__ out) {
    int j = blockIdx.x * blockDim.x + threadIdx.x;
    float p  = x[t * NX + (j >> 6)] * g[t * NG + (j & (NG - 1))];
    float hj = h[j];
    u[j] = p + hj;
    v[j] = YITA * (p - hj);
    float d = fabsf(p - hj);
    #pragma unroll
    for (int s = 16; s > 0; s >>= 1) d += __shfl_down(d, s, 32);
    if ((threadIdx.x & 31) == 0) atomicAdd(out, d);
}

// ---------------- M = lamda*M + outer(u, v)  (bf16, 8 elems/thread) ----------------
__global__ __launch_bounds__(256) void k_update_M(u16* __restrict__ Mb,
                                                  const float* __restrict__ u,
                                                  const float* __restrict__ v) {
    size_t i8   = (size_t)blockIdx.x * blockDim.x + threadIdx.x;
    size_t base = i8 * 8;
    int i = (int)(base >> 13);          // row (P = 2^13)
    int j = (int)(base & (P - 1));      // col
    uint4 m = *(uint4*)(Mb + base);
    float ui = u[i];
    float4 v0 = *(const float4*)(v + j);
    float4 v1 = *(const float4*)(v + j + 4);
    uint4 o;
    o.x = pair_update(m.x, ui, v0.x, v0.y);
    o.y = pair_update(m.y, ui, v0.z, v0.w);
    o.z = pair_update(m.z, ui, v1.x, v1.y);
    o.w = pair_update(m.w, ui, v1.z, v1.w);
    *(uint4*)(Mb + base) = o;
}

// ---------------- host orchestration ----------------
extern "C" void kernel_launch(void* const* d_in, const int* in_sizes, int n_in,
                              void* d_out, int out_size, void* d_ws, size_t ws_size,
                              hipStream_t stream) {
    const float* x  = (const float*)d_in[0];   // [T, NX]
    const float* g  = (const float*)d_in[1];   // [T, NG]
    const float* M0 = (const float*)d_in[2];   // [P, P] fp32

    char* ws = (char*)d_ws;
    u16*   Mb    = (u16*)ws;                 size_t off = (size_t)P * P * sizeof(u16);
    off += (size_t)1 << 20;                  // 1 MB pad: absorbs trailing OOB stage
    float* h     = (float*)(ws + off);       off += (size_t)P * sizeof(float);
    u16*   hbf   = (u16*)(ws + off);         off += (size_t)P * sizeof(u16);
    float* ypart = (float*)(ws + off);       off += (size_t)KSPLIT * P * sizeof(float);
    float* u     = (float*)(ws + off);       off += (size_t)P * sizeof(float);
    float* v     = (float*)(ws + off);       off += (size_t)P * sizeof(float);
    float* out   = (float*)d_out;

    (void)hipMemsetAsync(out, 0, sizeof(float), stream);

    // one-time: bf16 working copy of M (128 MB -> fits 192 MB L2)
    k_cvt<<<(int)(((size_t)P * P / 8) / 256), 256, 0, stream>>>(M0, Mb);

    dim3 ggrid(P / 128, KSPLIT);   // 64 x 8 = 512 blocks, 4096 waves
    for (int t = 0; t < T; ++t) {
        k_init_h<<<P / 256, 256, 0, stream>>>(g, t, h, hbf);
        for (int it = 0; it < N_ITER; ++it) {
            k_gemv<<<ggrid, 256, 0, stream>>>(Mb, hbf, ypart);
            k_update_h<<<P / 256, 256, 0, stream>>>(h, hbf, ypart);
        }
        k_finalize<<<P / 256, 256, 0, stream>>>(x, g, t, h, u, v, out);
        k_update_M<<<(int)(((size_t)P * P / 8) / 256), 256, 0, stream>>>(Mb, u, v);
    }
}
